// SSIM_13700945674826
// MI455X (gfx1250) — matrix operand order repaired and checked by reading, not compiled, not run on hardware
//
#include <hip/hip_runtime.h>

typedef __attribute__((ext_vector_type(16))) _Float16 v16h;
typedef __attribute__((ext_vector_type(8)))  float    v8f;

#define IMG_H 512
#define IMG_W 512
#define NB 16
#define NC 3
#define WAVES_PER_BLOCK 8
#define PATCH 20   // 16 + 2-halo each side

// One wave32 computes one 16x16 output tile of the channel-mean SSIM map.
// Vertical Gaussian pass is done as V_WMMA_F32_16X16X32_F16 with a banded
// A-matrix of taps; horizontal pass + products are fp32 VALU from LDS.
__global__ __launch_bounds__(256)
void ssim_wmma_kernel(const float* __restrict__ img1,
                      const float* __restrict__ img2,
                      const float* __restrict__ window,
                      float* __restrict__ out)
{
    __shared__ float lds1[WAVES_PER_BLOCK][PATCH * PATCH];
    __shared__ float lds2[WAVES_PER_BLOCK][PATCH * PATCH];

    const int tid  = threadIdx.x;
    const int wave = tid >> 5;
    const int lane = tid & 31;
    const int hi   = lane >> 4;     // 0: lanes 0-15, 1: lanes 16-31
    const int nm   = lane & 15;     // N column (B/D) and M row (A)

    // work unit: (batch, 16x16 tile). 16*32*32 = 16384 units, exact grid fit.
    const int unit = blockIdx.x * WAVES_PER_BLOCK + wave;
    const int b    = unit >> 10;
    const int t    = unit & 1023;
    const int r0   = (t >> 5) * 16;
    const int c0   = (t & 31) * 16;

    // Recover separable 1-D gaussian from the 5x5 window (w2d = outer(g,g)):
    // g[i] = w2d[2][i] / sqrt(w2d[2][2]); window layout [C,1,5,5], channel 0.
    float g[5];
    {
        float g2 = sqrtf(window[2 * 5 + 2]);
        #pragma unroll
        for (int i = 0; i < 5; ++i) g[i] = window[2 * 5 + i] / g2;
    }

    // A fragment (16x32 f16, ISA layout): banded vertical-conv matrix.
    // A[m][k] = g[k-m] for k in [m, m+4], k < 20; else 0.
    // Layout: lanes hold M = lane&15; VGPR i K-pairs: i<4 -> {2i,2i+1}(+8 hi),
    //         i>=4 -> {16+2(i-4), ...}(+8 hi).
    v16h af;
    #pragma unroll
    for (int i = 0; i < 8; ++i) {
        int kbase = ((i < 4) ? 2 * i : 16 + 2 * (i - 4)) + (hi ? 8 : 0);
        #pragma unroll
        for (int s = 0; s < 2; ++s) {
            int k = kbase + s;
            int d = k - nm;
            float v = (k < PATCH && d >= 0 && d <= 4) ? g[d] : 0.0f;
            af[2 * i + s] = (_Float16)v;
        }
    }

    const float C1v = 0.01f * 0.01f;
    const float C2v = 0.03f * 0.03f;
    v8f acc = {};   // channel-accumulated SSIM in D-matrix layout

    for (int ch = 0; ch < NC; ++ch) {
        const size_t base = (size_t)(b * NC + ch) * (size_t)(IMG_H * IMG_W);
        float* p1 = lds1[wave];
        float* p2 = lds2[wave];

        // 20x20 zero-padded patches (SAME padding) into LDS
        for (int idx = lane; idx < PATCH * PATCH; idx += 32) {
            int pr = idx / PATCH, pc = idx - pr * PATCH;
            int gr = r0 + pr - 2, gc = c0 + pc - 2;
            float a = 0.0f, c = 0.0f;
            if ((unsigned)gr < IMG_H && (unsigned)gc < IMG_W) {
                size_t off = base + (size_t)gr * IMG_W + gc;
                a = img1[off];
                c = img2[off];
            }
            p1[idx] = a;
            p2[idx] = c;
        }
        __syncthreads();

        // Horizontal 5-tap pass (fp32) -> B fragments (32x16 f16, ISA layout):
        // lanes hold N = lane&15; VGPR i K-pairs: i<4 -> {2i,2i+1}, i>=4 -> {16+2(i-4), ...} (+8 for high lanes).
        // Valid H rows: K=0..19 (rows r0-2 .. r0+17); K>=20 padded zero.
        v16h b1, b2, b11, b22, b12;
        #pragma unroll
        for (int i = 0; i < 8; ++i) {
            int kbase = ((i < 4) ? 2 * i : 16 + 2 * (i - 4)) + (hi ? 8 : 0);
            #pragma unroll
            for (int s = 0; s < 2; ++s) {
                int k = kbase + s;
                float h1 = 0.f, h2 = 0.f, h11 = 0.f, h22 = 0.f, h12 = 0.f;
                if (k < PATCH) {
                    const float* row1 = p1 + k * PATCH + nm;
                    const float* row2 = p2 + k * PATCH + nm;
                    #pragma unroll
                    for (int j = 0; j < 5; ++j) {
                        float w = g[j];
                        float a = row1[j];
                        float c = row2[j];
                        h1  += w * a;
                        h2  += w * c;
                        h11 += w * a * a;
                        h22 += w * c * c;
                        h12 += w * a * c;
                    }
                }
                int e = 2 * i + s;
                b1[e]  = (_Float16)h1;
                b2[e]  = (_Float16)h2;
                b11[e] = (_Float16)h11;
                b22[e] = (_Float16)h22;
                b12[e] = (_Float16)h12;
            }
        }

        // Vertical pass: one WMMA per map (fp16 in, fp32 accumulate)
        v8f z = {};
        v8f mu1 = __builtin_amdgcn_wmma_f32_16x16x32_f16(false, af, false, b1,  (short)0, z, false, false);
        v8f mu2 = __builtin_amdgcn_wmma_f32_16x16x32_f16(false, af, false, b2,  (short)0, z, false, false);
        v8f e11 = __builtin_amdgcn_wmma_f32_16x16x32_f16(false, af, false, b11, (short)0, z, false, false);
        v8f e22 = __builtin_amdgcn_wmma_f32_16x16x32_f16(false, af, false, b22, (short)0, z, false, false);
        v8f e12 = __builtin_amdgcn_wmma_f32_16x16x32_f16(false, af, false, b12, (short)0, z, false, false);

        // Elementwise SSIM on the D fragment, accumulate channel mean
        #pragma unroll
        for (int j = 0; j < 8; ++j) {
            float m1 = mu1[j], m2 = mu2[j];
            float m1s = m1 * m1, m2s = m2 * m2, m12 = m1 * m2;
            float s1  = e11[j] - m1s;
            float s2  = e22[j] - m2s;
            float s12 = e12[j] - m12;
            float num = (2.0f * m12 + C1v) * (2.0f * s12 + C2v);
            float den = (m1s + m2s + C1v) * (s1 + s2 + C2v);
            acc[j] += num / den;
        }
        __syncthreads();   // LDS patch reuse next channel
    }

    // Store mean over channels. D layout: row = j + 8*hi, col = lane&15.
    const size_t obase = (size_t)b * (size_t)(IMG_H * IMG_W);
    #pragma unroll
    for (int j = 0; j < 8; ++j) {
        int row = r0 + j + 8 * hi;
        int col = c0 + nm;
        out[obase + (size_t)row * IMG_W + col] = acc[j] * (1.0f / 3.0f);
    }
}

extern "C" void kernel_launch(void* const* d_in, const int* in_sizes, int n_in,
                              void* d_out, int out_size, void* d_ws, size_t ws_size,
                              hipStream_t stream) {
    const float* img1   = (const float*)d_in[0];
    const float* img2   = (const float*)d_in[1];
    const float* window = (const float*)d_in[2];
    float* out = (float*)d_out;

    // 16 batches * (32*32) tiles = 16384 waves; 8 waves (256 thr) per block
    const int units  = NB * (IMG_H / 16) * (IMG_W / 16);
    const int blocks = units / WAVES_PER_BLOCK;   // 2048
    hipLaunchKernelGGL(ssim_wmma_kernel, dim3(blocks), dim3(256), 0, stream,
                       img1, img2, window, out);
}
